// Model_39960375722226
// MI455X (gfx1250) — matrix operand order repaired and checked by reading, not compiled, not run on hardware
//
#include <hip/hip_runtime.h>
#include <hip/hip_bf16.h>
#include <math.h>

// ---------------------------------------------------------------------------
// VQ-VAE forward for gfx1250 (MI455X).
//  - All deep-channel convs -> im2col WMMA GEMM (v_wmma_f32_16x16x32_f16).
//  - Intermediates NHWC; pre-VQ conv output IS the flat [M,D] VQ matrix and
//    the codebook-distance GEMM reuses the same kernel.
//  - Weights are repacked once to f16 B-layout [N][Kc] (k = (kh,kw,ic)) so
//    B tiles are contiguous and streamed with GLOBAL_LOAD_ASYNC_TO_LDS_B128
//    (ASYNCcnt) — builtin confirmed present on this toolchain; params are
//    V4i-typed pointers (addrspace(1) src, addrspace(3) dst, imm, imm cpol).
// ---------------------------------------------------------------------------

typedef __attribute__((ext_vector_type(16))) _Float16 v16h;
typedef __attribute__((ext_vector_type(8)))  float    v8f;

union FragH { v16h h; unsigned u[8]; };

#if __has_builtin(__builtin_amdgcn_global_load_async_to_lds_b128)
#define HAVE_ASYNC_LDS 1
// exact pointee type of the builtin's pointer params: GCC-vector int4
typedef int v4i_gcc __attribute__((vector_size(16)));
typedef __attribute__((address_space(1))) v4i_gcc* g_v4i_p;
typedef __attribute__((address_space(3))) v4i_gcc* l_v4i_p;
#else
#define HAVE_ASYNC_LDS 0
#endif

__device__ __forceinline__ void wait_async_lds()
{
#if __has_builtin(__builtin_amdgcn_s_wait_asynccnt)
    __builtin_amdgcn_s_wait_asynccnt(0);
#else
    asm volatile("s_wait_asynccnt 0" ::: "memory");
#endif
}

// ---------------------------------------------------------------------------
// Weight repack: OIHW f32 -> f16 B-layout [N][Kc], Kc = CIN*KH*KW,
// k = (kh*KW + kw)*CIN + ic  (ic fastest, matches NHWC im2col A tiles).
// ---------------------------------------------------------------------------
__global__ void repack_w(const float* __restrict__ src, _Float16* __restrict__ dst,
                         int total, int CIN, int KH, int KW)
{
    int i = blockIdx.x * blockDim.x + threadIdx.x;
    if (i >= total) return;
    int kw = i % KW; int r = i / KW;
    int kh = r % KH; r /= KH;
    int ic = r % CIN; int n = r / CIN;
    dst[(size_t)n * CIN * KH * KW + (kh * KW + kw) * CIN + ic] = (_Float16)src[i];
}

// ---------------------------------------------------------------------------
// conv-as-GEMM, WMMA tiled. C[M,N], M = B*OH*OW, N = Cout.
// Tile BM=128, BN=64, BK=32; 8 waves, each owns a 32x32 patch (2x2 WMMA).
// Spatial geometry is compile-time -> all index math is shifts/masks.
// ---------------------------------------------------------------------------
template <int CIN, int KH, int KW, int S, int P, int IH, int IW, int OH, int OW,
          bool RELU_IN, bool RELU_OUT, bool ADD_RES, bool HAS_BIAS>
__global__ __launch_bounds__(256) void conv_wmma(
    const float* __restrict__ in,      // NHWC [B, IH, IW, CIN]
    const _Float16* __restrict__ wgt,  // packed [N][Kc]
    const float* __restrict__ bias,    // [N] or null
    const float* __restrict__ res,     // [M, N] or null
    float* __restrict__ out,           // [M, N]
    int BATCH, int N)
{
    constexpr int Kc = CIN * KH * KW;
    const int M = BATCH * OH * OW;

    const int tid  = threadIdx.x;
    const int lane = tid & 31;
    const int wid  = tid >> 5;
    const int wm   = wid & 3;
    const int wn   = wid >> 2;

    const int blockM = blockIdx.x * 128;
    const int blockN = blockIdx.y * 64;

    __shared__ __align__(16) _Float16 sA[128][40];  // padded rows
    __shared__ __align__(16) _Float16 sB[64][40];

    v8f c[2][2] = {};

    if (tid == 0) __builtin_prefetch(in, 0, 0);  // global_prefetch_b8 hint

    for (int k0 = 0; k0 < Kc; k0 += 32) {
#if HAVE_ASYNC_LDS
        // ---- async-stage B tile: 64 rows x 64B; one b128/lane, 8 rows/wave
        {
            int nn  = wid * 8 + (lane >> 2);
            int seg = lane & 3;
            if (blockN + nn < N) {
                const _Float16* src =
                    wgt + (size_t)(blockN + nn) * Kc + k0 + seg * 8;
                __builtin_amdgcn_global_load_async_to_lds_b128(
                    (g_v4i_p)src, (l_v4i_p)&sB[nn][seg * 8], 0, 0);
            } else {
                *(uint4*)&sB[nn][seg * 8] = make_uint4(0u, 0u, 0u, 0u);
            }
        }
#else
        #pragma unroll
        for (int i = 0; i < 8; ++i) {
            int idx = i * 256 + tid;
            int nn = idx >> 5, kk = idx & 31;
            int ng = blockN + nn;
            sB[nn][kk] = (ng < N) ? wgt[(size_t)ng * Kc + k0 + kk]
                                  : (_Float16)0.f;
        }
#endif
        // ---- stage A tile: 128x32 f16 im2col, 16 elements/thread ----
        #pragma unroll
        for (int i = 0; i < 16; ++i) {
            int idx = i * 256 + tid;
            int mm = idx >> 5, kk = idx & 31;
            int mg = blockM + mm;
            int kg = k0 + kk;
            int ic = kg % CIN;
            int r  = kg / CIN;
            int kw = r % KW;
            int kh = r / KW;
            float v = 0.f;
            if (mg < M) {
                int ox = mg % OW; int t = mg / OW;   // OW/OH const -> shifts
                int oy = t % OH;  int b = t / OH;
                int iy = oy * S - P + kh;
                int ix = ox * S - P + kw;
                if (iy >= 0 && iy < IH && ix >= 0 && ix < IW)
                    v = in[(((size_t)b * IH + iy) * IW + ix) * CIN + ic];
                if (RELU_IN) v = v > 0.f ? v : 0.f;
            }
            sA[mm][kk] = (_Float16)v;
        }
#if HAVE_ASYNC_LDS
        wait_async_lds();
#endif
        __syncthreads();

        // ---- fragments (contiguous k = 16*half + s, same for A and B) ----
        const int hi8  = (lane >> 4) << 3;   // unused; A also uses hi16 (+16 K)
        const int hi16 = (lane >> 4) << 4;   // B: +16 K on hi half-wave
        const int row  = lane & 15;

        FragH a[2], b[2];
        #pragma unroll
        for (int ti = 0; ti < 2; ++ti) {
            int ml = wm * 32 + ti * 16 + row;
            #pragma unroll
            for (int p = 0; p < 8; ++p) {
                int kp = 2 * p + hi16;
                a[ti].u[p] = *(const unsigned*)&sA[ml][kp];
            }
        }
        #pragma unroll
        for (int tj = 0; tj < 2; ++tj) {
            int nl = wn * 32 + tj * 16 + row;
            #pragma unroll
            for (int p = 0; p < 8; ++p) {
                int kp = 2 * p + hi16;
                b[tj].u[p] = *(const unsigned*)&sB[nl][kp];
            }
        }

        #pragma unroll
        for (int ti = 0; ti < 2; ++ti)
            #pragma unroll
            for (int tj = 0; tj < 2; ++tj)
                c[ti][tj] = __builtin_amdgcn_wmma_f32_16x16x32_f16(
                    false, a[ti].h, false, b[tj].h,
                    (short)0, c[ti][tj], false, false);

        __syncthreads();
    }

    // ---- epilogue (C/D: elem e -> row e+8*hiLane, col lane&15) ----
    #pragma unroll
    for (int ti = 0; ti < 2; ++ti) {
        #pragma unroll
        for (int tj = 0; tj < 2; ++tj) {
            int ng = blockN + wn * 32 + tj * 16 + (lane & 15);
            #pragma unroll
            for (int e = 0; e < 8; ++e) {
                int mg = blockM + wm * 32 + ti * 16 + e + ((lane >> 4) << 3);
                if (mg < M && ng < N) {
                    float v = c[ti][tj][e];
                    if (HAS_BIAS) v += bias[ng];
                    if (ADD_RES)  v += res[(size_t)mg * N + ng];
                    if (RELU_OUT) v = v > 0.f ? v : 0.f;
                    out[(size_t)mg * N + ng] = v;
                }
            }
        }
    }
}

// ---------------------------------------------------------------------------
// First conv: 3 input channels (K=48) -> direct. x NCHW -> out NHWC + ReLU.
// ---------------------------------------------------------------------------
__global__ __launch_bounds__(256) void conv1_k(
    const float* __restrict__ x, const float* __restrict__ w,
    const float* __restrict__ bias, float* __restrict__ out)
{
    size_t gid = (size_t)blockIdx.x * blockDim.x + threadIdx.x;
    int oc = gid & 63;  size_t r = gid >> 6;
    int ox = r & 127;   r >>= 7;
    int oy = r & 127;   int b = (int)(r >> 7);
    float acc = bias[oc];
    #pragma unroll
    for (int ic = 0; ic < 3; ++ic)
        #pragma unroll
        for (int kh = 0; kh < 4; ++kh) {
            int iy = oy * 2 - 1 + kh;
            if (iy < 0 || iy >= 256) continue;
            #pragma unroll
            for (int kw = 0; kw < 4; ++kw) {
                int ix = ox * 2 - 1 + kw;
                if (ix < 0 || ix >= 256) continue;
                acc += x[(((size_t)b * 3 + ic) * 256 + iy) * 256 + ix] *
                       w[((oc * 3 + ic) * 4 + kh) * 4 + kw];
            }
        }
    out[gid] = acc > 0.f ? acc : 0.f;
}

// ---------------------------------------------------------------------------
// ConvTranspose2d(k=4,s=2,p=1): 4 taps x CIN MACs/output -> direct.
// ---------------------------------------------------------------------------
template <int CIN, int COUT, bool RELU_IN, bool RELU_OUT, bool OUT_NCHW>
__global__ __launch_bounds__(256) void tconv4x4s2(
    const float* __restrict__ in, const float* __restrict__ wgt,
    const float* __restrict__ bias, float* __restrict__ out,
    int B, int IH, int IW)
{
    int OH = IH * 2, OW = IW * 2;
    size_t total = (size_t)B * OH * OW * COUT;
    size_t gid = (size_t)blockIdx.x * blockDim.x + threadIdx.x;
    if (gid >= total) return;
    int oc = (int)(gid % COUT); size_t r = gid / COUT;
    int ox = (int)(r % OW); r /= OW;
    int oy = (int)(r % OH); int b = (int)(r / OH);

    float acc = bias[oc];
    #pragma unroll
    for (int kh = 0; kh < 4; ++kh) {
        int y2 = oy + kh - 2;
        if (y2 < 0 || (y2 & 1)) continue;
        int iy = y2 >> 1; if (iy >= IH) continue;
        #pragma unroll
        for (int kw = 0; kw < 4; ++kw) {
            int x2 = ox + kw - 2;
            if (x2 < 0 || (x2 & 1)) continue;
            int ix = x2 >> 1; if (ix >= IW) continue;
            const float* ip = in + (((size_t)b * IH + iy) * IW + ix) * CIN;
            const float* wp = wgt + ((size_t)oc * CIN * 4 + kh) * 4 + kw;
            for (int ic = 0; ic < CIN; ++ic) {
                float v = ip[ic];
                if (RELU_IN) v = v > 0.f ? v : 0.f;
                acc += v * wp[(size_t)ic * 16];
            }
        }
    }
    if (RELU_OUT) acc = acc > 0.f ? acc : 0.f;
    if (OUT_NCHW)
        out[(((size_t)b * COUT + oc) * OH + oy) * OW + ox] = acc;
    else
        out[(((size_t)b * OH + oy) * OW + ox) * COUT + oc] = acc;
}

// ---------------------------------------------------------------------------
// VQ pieces
// ---------------------------------------------------------------------------
__global__ void enorm_k(const float* __restrict__ E, float* __restrict__ en,
                        int K, int D)
{
    int k = blockIdx.x * blockDim.x + threadIdx.x;
    if (k >= K) return;
    const float* r = E + (size_t)k * D;
    float s = 0.f;
    for (int d = 0; d < D; ++d) s += r[d] * r[d];
    en[k] = s;
}

__global__ void vq_argmin(const float* __restrict__ dots,
                          const float* __restrict__ en,
                          int* __restrict__ idx, int M, int K)
{
    int m = blockIdx.x * blockDim.x + threadIdx.x;
    if (m >= M) return;
    const float* row = dots + (size_t)m * K;
    float best = 3.4e38f; int bi = 0;
    for (int k = 0; k < K; ++k) {
        float d = en[k] - 2.f * row[k];
        if (d < best) { best = d; bi = k; }    // first-min, matches argmin
    }
    idx[m] = bi;
}

__global__ void zero_k(float* p, int n)
{
    int i = blockIdx.x * blockDim.x + threadIdx.x;
    if (i < n) p[i] = 0.f;
}

__global__ __launch_bounds__(256) void vq_gather(
    const float* __restrict__ z, const float* __restrict__ E,
    const int* __restrict__ idx, float* __restrict__ q,
    float* __restrict__ counts, float* __restrict__ loss_sum, int M, int D)
{
    int m = blockIdx.x * (blockDim.x >> 5) + (threadIdx.x >> 5);
    int lane = threadIdx.x & 31;
    if (m >= M) return;
    int k = idx[m];
    const float* zr = z + (size_t)m * D;
    const float* er = E + (size_t)k * D;
    float*       qr = q + (size_t)m * D;
    float s = 0.f;
    for (int d = lane; d < D; d += 32) {
        float e = er[d];
        float dz = e - zr[d];
        qr[d] = e;
        s += dz * dz;
    }
    #pragma unroll
    for (int off = 16; off > 0; off >>= 1) s += __shfl_down(s, off, 32);
    if (lane == 0) {
        atomicAdd(loss_sum, s);
        atomicAdd(&counts[k], 1.0f);
    }
}

__global__ void vq_finalize(const float* __restrict__ counts,
                            const float* __restrict__ loss_sum,
                            float* __restrict__ out_loss,
                            float* __restrict__ out_perp,
                            float invMD, float invM)
{
    __shared__ float sh[512];
    int t = threadIdx.x;
    float avg = counts[t] * invM;
    sh[t] = avg * logf(avg + 1e-10f);
    __syncthreads();
    for (int s = 256; s > 0; s >>= 1) {
        if (t < s) sh[t] += sh[t + s];
        __syncthreads();
    }
    if (t == 0) {
        *out_perp = expf(-sh[0]);
        *out_loss = 1.25f * (*loss_sum) * invMD;   // q_loss + 0.25*e_loss
    }
}

// ---------------------------------------------------------------------------
// Host orchestration
// ---------------------------------------------------------------------------
extern "C" void kernel_launch(void* const* d_in, const int* in_sizes, int n_in,
                              void* d_out, int out_size, void* d_ws, size_t ws_size,
                              hipStream_t stream)
{
    (void)in_sizes; (void)n_in; (void)ws_size;
    const float* x     = (const float*)d_in[0];
    const float* ew1   = (const float*)d_in[1];
    const float* eb1   = (const float*)d_in[2];
    const float* ew2   = (const float*)d_in[3];
    const float* eb2   = (const float*)d_in[4];
    const float* ew3   = (const float*)d_in[5];
    const float* eb3   = (const float*)d_in[6];
    const float* er0w1 = (const float*)d_in[7];
    const float* er0w2 = (const float*)d_in[8];
    const float* er1w1 = (const float*)d_in[9];
    const float* er1w2 = (const float*)d_in[10];
    const float* pw    = (const float*)d_in[11];
    const float* pb    = (const float*)d_in[12];
    const float* emb   = (const float*)d_in[13];
    const float* dw1   = (const float*)d_in[14];
    const float* db1   = (const float*)d_in[15];
    const float* dr0w1 = (const float*)d_in[16];
    const float* dr0w2 = (const float*)d_in[17];
    const float* dr1w1 = (const float*)d_in[18];
    const float* dr1w2 = (const float*)d_in[19];
    const float* tw1   = (const float*)d_in[20];
    const float* tb1   = (const float*)d_in[21];
    const float* tw2   = (const float*)d_in[22];
    const float* tb2   = (const float*)d_in[23];

    float* out = (float*)d_out;

    float* ws = (float*)d_ws;
    size_t off = 0;
    auto alloc = [&](size_t n) {
        float* p = ws + off;
        off += (n + 63) & ~(size_t)63;
        return p;
    };
    float* h1     = alloc(16777216);   // [16,128,128,64] NHWC
    float* h2     = alloc(8388608);    // [16,64,64,128]  NHWC
    float* h3     = alloc(8388608);    // [16,64,64,128]  NHWC
    float* rtmp   = alloc(2097152);    // [16,64,64,32]   NHWC
    float* z      = alloc(33554432);   // flat [65536,512]
    float* dots   = alloc(33554432);   // [65536,512] -> reused as q
    float* en     = alloc(512);
    float* counts = alloc(1024);       // counts[512] + loss_sum
    int*   idx    = (int*)alloc(65536);
    float* lsum   = counts + 512;
    float* q      = dots;
    float* d1     = h2;                // dead regions reused
    float* t1     = h1;

    // f16 packed-weight pool
    _Float16* hb = (_Float16*)alloc(700000);
    size_t hoff = 0;
    auto halloc = [&](size_t n) {
        _Float16* p = hb + hoff;
        hoff += (n + 63) & ~(size_t)63;
        return p;
    };
    _Float16* ew2h   = halloc(131072);
    _Float16* ew3h   = halloc(147456);
    _Float16* er0w1h = halloc(36864);
    _Float16* er0w2h = halloc(4096);
    _Float16* er1w1h = halloc(36864);
    _Float16* er1w2h = halloc(4096);
    _Float16* pwh    = halloc(65536);
    _Float16* embh   = halloc(262144);
    _Float16* dw1h   = halloc(589824);
    _Float16* dr0w1h = halloc(36864);
    _Float16* dr0w2h = halloc(4096);
    _Float16* dr1w1h = halloc(36864);
    _Float16* dr1w2h = halloc(4096);

    const dim3 blk(256);
    auto repack = [&](const float* w, _Float16* dst, int N_, int C_, int KH_, int KW_) {
        int tot = N_ * C_ * KH_ * KW_;
        repack_w<<<(tot + 255) / 256, blk, 0, stream>>>(w, dst, tot, C_, KH_, KW_);
    };
    repack(ew2,   ew2h,   128,  64, 4, 4);
    repack(ew3,   ew3h,   128, 128, 3, 3);
    repack(er0w1, er0w1h,  32, 128, 3, 3);
    repack(er0w2, er0w2h, 128,  32, 1, 1);
    repack(er1w1, er1w1h,  32, 128, 3, 3);
    repack(er1w2, er1w2h, 128,  32, 1, 1);
    repack(pw,    pwh,    512, 128, 1, 1);
    repack(emb,   embh,   512, 512, 1, 1);
    repack(dw1,   dw1h,   128, 512, 3, 3);
    repack(dr0w1, dr0w1h,  32, 128, 3, 3);
    repack(dr0w2, dr0w2h, 128,  32, 1, 1);
    repack(dr1w1, dr1w1h,  32, 128, 3, 3);
    repack(dr1w2, dr1w2h, 128,  32, 1, 1);

    // ---- Encoder ----
    conv1_k<<<65536, blk, 0, stream>>>(x, ew1, eb1, h1);

    conv_wmma<64, 4, 4, 2, 1, 128, 128, 64, 64, false, true, false, true>
        <<<dim3(512, 2), blk, 0, stream>>>(h1, ew2h, eb2, nullptr, h2, 16, 128);
    conv_wmma<128, 3, 3, 1, 1, 64, 64, 64, 64, false, false, false, true>
        <<<dim3(512, 2), blk, 0, stream>>>(h2, ew3h, eb3, nullptr, h3, 16, 128);
    // resblock 0 (in-place on h3)
    conv_wmma<128, 3, 3, 1, 1, 64, 64, 64, 64, true, false, false, false>
        <<<dim3(512, 1), blk, 0, stream>>>(h3, er0w1h, nullptr, nullptr, rtmp, 16, 32);
    conv_wmma<32, 1, 1, 1, 0, 64, 64, 64, 64, true, false, true, false>
        <<<dim3(512, 2), blk, 0, stream>>>(rtmp, er0w2h, nullptr, h3, h3, 16, 128);
    // resblock 1
    conv_wmma<128, 3, 3, 1, 1, 64, 64, 64, 64, true, false, false, false>
        <<<dim3(512, 1), blk, 0, stream>>>(h3, er1w1h, nullptr, nullptr, rtmp, 16, 32);
    conv_wmma<32, 1, 1, 1, 0, 64, 64, 64, 64, true, false, true, false>
        <<<dim3(512, 2), blk, 0, stream>>>(rtmp, er1w2h, nullptr, h3, h3, 16, 128);
    // relu -> pre-VQ 1x1 (128->512): output IS flat z [65536,512]
    conv_wmma<128, 1, 1, 1, 0, 64, 64, 64, 64, true, false, false, true>
        <<<dim3(512, 8), blk, 0, stream>>>(h3, pwh, pb, nullptr, z, 16, 512);

    // ---- VQ ----
    enorm_k<<<2, blk, 0, stream>>>(emb, en, 512, 512);
    conv_wmma<512, 1, 1, 1, 0, 64, 64, 64, 64, false, false, false, false>
        <<<dim3(512, 8), blk, 0, stream>>>(z, embh, nullptr, nullptr, dots, 16, 512);
    vq_argmin<<<256, blk, 0, stream>>>(dots, en, idx, 65536, 512);
    zero_k<<<3, blk, 0, stream>>>(counts, 513);
    vq_gather<<<8192, blk, 0, stream>>>(z, emb, idx, q, counts, lsum, 65536, 512);
    vq_finalize<<<1, 512, 0, stream>>>(counts, lsum,
                                       out, out + (out_size - 1),
                                       1.0f / (65536.0f * 512.0f),
                                       1.0f / 65536.0f);

    // ---- Decoder ---- (quant == q numerically; straight-through)
    conv_wmma<512, 3, 3, 1, 1, 64, 64, 64, 64, false, false, false, true>
        <<<dim3(512, 2), blk, 0, stream>>>(q, dw1h, db1, nullptr, d1, 16, 128);
    conv_wmma<128, 3, 3, 1, 1, 64, 64, 64, 64, true, false, false, false>
        <<<dim3(512, 1), blk, 0, stream>>>(d1, dr0w1h, nullptr, nullptr, rtmp, 16, 32);
    conv_wmma<32, 1, 1, 1, 0, 64, 64, 64, 64, true, false, true, false>
        <<<dim3(512, 2), blk, 0, stream>>>(rtmp, dr0w2h, nullptr, d1, d1, 16, 128);
    conv_wmma<128, 3, 3, 1, 1, 64, 64, 64, 64, true, false, false, false>
        <<<dim3(512, 1), blk, 0, stream>>>(d1, dr1w1h, nullptr, nullptr, rtmp, 16, 32);
    conv_wmma<32, 1, 1, 1, 0, 64, 64, 64, 64, true, false, true, false>
        <<<dim3(512, 2), blk, 0, stream>>>(rtmp, dr1w2h, nullptr, d1, d1, 16, 128);

    // relu -> tconv1 (128->64) + relu, then tconv2 (64->3) NCHW into d_out
    tconv4x4s2<128, 64, true, true, false>
        <<<65536, blk, 0, stream>>>(d1, tw1, tb1, t1, 16, 64, 64);
    tconv4x4s2<64, 3, false, false, true>
        <<<12288, blk, 0, stream>>>(t1, tw2, tb2, out + 1, 16, 128, 128);
}
